// Theta2TResNet_ClassHead_81252191306034
// MI455X (gfx1250) — matrix operand order repaired and checked by reading, not compiled, not run on hardware
//
#include <hip/hip_runtime.h>
#include <hip/hip_bf16.h>
#include <stdint.h>

typedef __bf16 bf16;
typedef __attribute__((ext_vector_type(16))) __bf16 v16bf;
typedef __attribute__((ext_vector_type(8)))  float   v8f;

#define B_TOT   16384
#define THETA   64
#define YDIM    16
#define OUTD    256
#define WIDTH   1024
#define HID     2048
#define NBLK    4
#define KIN_PAD 96
#define KIN_REAL 81

// ---------------------------------------------------------------------------
// helpers
// ---------------------------------------------------------------------------
__device__ __forceinline__ bf16 f2bf(float f) {
  union { float f; uint32_t u; } x; x.f = f;
  uint32_t r = x.u + 0x7FFFu + ((x.u >> 16) & 1u);
  uint16_t h = (uint16_t)(r >> 16);
  return __builtin_bit_cast(bf16, h);
}

__device__ __forceinline__ v8f wmma_bf16(v16bf a, v16bf b, v8f c) {
  // D = A(16x32 bf16) * B(32x16 bf16) + C(16x16 f32)
  return __builtin_amdgcn_wmma_f32_16x16x32_bf16(
      false, a, false, b, (short)0, c, false, false);
}

// A fragment: 16x32 bf16 from row-major source. lane m = lane&15 holds
// K = kb..kb+7 (elems 0..7) and kb+16..kb+23 (elems 8..15), kb = (lane<16)?0:8
__device__ __forceinline__ v16bf load_frag_a(const bf16* __restrict__ Arow0,
                                             int lda, int k0, int lane) {
  int m  = lane & 15;
  int kb = (lane >> 4) << 3;
  const bf16* p = Arow0 + (size_t)m * lda + (k0 + kb);
  union { v16bf v; uint4 q[2]; } u;
  u.q[0] = *(const uint4*)(p);
  u.q[1] = *(const uint4*)(p + 16);
  return u.v;
}

// B fragment: 512 bf16 per 32x16 tile, fragment-linear; lane reads 16
// contiguous elements (two b128 loads -> ds_load_b128 when src is LDS).
__device__ __forceinline__ v16bf load_frag_b(const bf16* fragbase, int lane) {
  const bf16* p = fragbase + lane * 16;
  union { v16bf v; uint4 q[2]; } u;
  u.q[0] = *(const uint4*)(p);
  u.q[1] = *(const uint4*)(p + 8);
  return u.v;
}

// CDNA5 async DMA: global -> LDS, 16 bytes, tracked by ASYNCcnt.
// LDS offset = low 32 bits of the generic pointer (LDS aperture truncation).
__device__ __forceinline__ void async_stage_b128(const bf16* g, bf16* l) {
  uint32_t loff = (uint32_t)(uintptr_t)(void*)l;
  asm volatile("global_load_async_to_lds_b128 %0, %1, off"
               :: "v"(loff), "v"((uint64_t)(uintptr_t)g)
               : "memory");
}
__device__ __forceinline__ void async_wait_barrier() {
  asm volatile("s_wait_asynccnt 0x0" ::: "memory");
  __syncthreads();
}

// ---------------------------------------------------------------------------
// prep kernels
// ---------------------------------------------------------------------------
__global__ void build_x_kernel(const float* __restrict__ theta,
                               const float* __restrict__ lam,
                               const float* __restrict__ yoh,
                               bf16* __restrict__ x) {
  int idx = blockIdx.x * blockDim.x + threadIdx.x;
  if (idx >= B_TOT * KIN_PAD) return;
  int b = idx / KIN_PAD, c = idx - b * KIN_PAD;
  float v;
  if (c < THETA)         v = theta[b * THETA + c];
  else if (c == THETA)   v = lam[b];
  else if (c < KIN_REAL) v = yoh[b * YDIM + (c - THETA - 1)];
  else                   v = 0.0f;
  x[idx] = f2bf(v);
}

__global__ void zero16_kernel(int* __restrict__ cnt) {
  if (threadIdx.x < YDIM) cnt[threadIdx.x] = 0;
}

__global__ void argmax_bin_kernel(const float* __restrict__ yoh,
                                  int* __restrict__ cnt,
                                  int* __restrict__ rows) {
  int b = blockIdx.x * blockDim.x + threadIdx.x;
  if (b >= B_TOT) return;
  const float* y = yoh + (size_t)b * YDIM;
  int best = 0; float bv = y[0];
  #pragma unroll
  for (int j = 1; j < YDIM; ++j) { float v = y[j]; if (v > bv) { bv = v; best = j; } }
  int pos = atomicAdd(&cnt[best], 1);
  rows[best * B_TOT + pos] = b;
}

// src f32 [realK, N] row-major -> dst bf16 [N/16][Kp/32][32 lanes][16 elems]
__global__ void shuffle_w_kernel(const float* __restrict__ src,
                                 bf16* __restrict__ dst,
                                 int realK, int Kp, int N) {
  int nk = Kp >> 5;
  long total = (long)nk * (N >> 4) * 512;
  long idx = (long)blockIdx.x * blockDim.x + threadIdx.x;
  if (idx >= total) return;
  int e  = (int)(idx & 15);
  int l  = (int)((idx >> 4) & 31);
  long rest = idx >> 9;
  int tk = (int)(rest % nk);
  int tn = (int)(rest / nk);
  int col = l & 15;
  int kb  = (l < 16) ? 0 : 8;
  int k = tk * 32 + kb + (e & 7) + ((e >> 3) << 4);
  int n = tn * 16 + col;
  float v = (k < realK) ? src[(size_t)k * N + n] : 0.0f;
  dst[idx] = f2bf(v);
}

// ---------------------------------------------------------------------------
// Staged GEMM skeleton: WG = 256 thr = 8 waves, tile 256 rows x 64 cols.
// Each wave: 32 rows x 64 cols (2x4 accumulators, 8 WMMAs / K-step).
// B slab (4 tiles x 512 bf16 = 4KB) double-buffered in LDS via async DMA.
// ---------------------------------------------------------------------------
#define GEMM_PROLOG(Aptr, lda_, Wfbase, nk_)                                   \
  __shared__ __align__(32) bf16 Bs[2][2048];                                   \
  const int tid  = threadIdx.x;                                                \
  const int lane = tid & 31;                                                   \
  const int wave = tid >> 5;                                                   \
  const int nk   = (nk_);                                                      \
  const int lda  = (lda_);                                                     \
  const int row0 = blockIdx.y * 256 + wave * 32;                               \
  const int col0 = blockIdx.x * 64;                                            \
  const int tn0  = blockIdx.x * 4;                                             \
  const int sidx = (tid >> 6) * 512 + (tid & 63) * 8;                          \
  const bf16* gsrc = (Wfbase) + ((size_t)(tn0 + (tid >> 6)) * nk) * 512 +      \
                     (size_t)(tid & 63) * 8;                                   \
  const bf16* A0 = (Aptr) + (size_t)row0 * lda;                                \
  const bf16* A1 = (Aptr) + (size_t)(row0 + 16) * lda;                         \
  v8f acc[2][4] = {};                                                          \
  async_stage_b128(gsrc, &Bs[0][sidx]);                                        \
  async_wait_barrier();

#define GEMM_KLOOP()                                                           \
  for (int t = 0; t < nk; ++t) {                                               \
    const int buf = t & 1;                                                     \
    if (t + 1 < nk)                                                            \
      async_stage_b128(gsrc + (size_t)(t + 1) * 512, &Bs[buf ^ 1][sidx]);      \
    int k0 = t << 5;                                                           \
    v16bf a0 = load_frag_a(A0, lda, k0, lane);                                 \
    v16bf a1 = load_frag_a(A1, lda, k0, lane);                                 \
    v16bf bfr[4];                                                              \
    _Pragma("unroll")                                                          \
    for (int ns = 0; ns < 4; ++ns) bfr[ns] = load_frag_b(&Bs[buf][ns * 512], lane); \
    _Pragma("unroll")                                                          \
    for (int ns = 0; ns < 4; ++ns) {                                           \
      acc[0][ns] = wmma_bf16(a0, bfr[ns], acc[0][ns]);                         \
      acc[1][ns] = wmma_bf16(a1, bfr[ns], acc[1][ns]);                         \
    }                                                                          \
    async_wait_barrier();                                                      \
  }

// --- input layer: K=96, N=1024, GELU epilogue, f32 out -----------------------
__global__ __launch_bounds__(256) void gemm_in_gelu_kernel(
    const bf16* __restrict__ X, const bf16* __restrict__ Wf,
    const float* __restrict__ bias, float* __restrict__ H) {
  GEMM_PROLOG(X, KIN_PAD, Wf, KIN_PAD >> 5)
  GEMM_KLOOP()
  const int colL = lane & 15;
  const int mB   = (lane < 16) ? 0 : 8;
  #pragma unroll
  for (int ms = 0; ms < 2; ++ms)
    #pragma unroll
    for (int ns = 0; ns < 4; ++ns) {
      int col = col0 + ns * 16 + colL;
      float bv = bias[col];
      #pragma unroll
      for (int r = 0; r < 8; ++r) {
        int row = row0 + ms * 16 + mB + r;
        float v = acc[ms][ns][r] + bv;
        v = 0.5f * v * (1.0f + erff(v * 0.70710678118f));
        H[(size_t)row * WIDTH + col] = v;
      }
    }
}

// --- z @ w1 + b1, SiLU, bf16 out: K=1024, N=2048 -----------------------------
__global__ __launch_bounds__(256) void gemm_silu_kernel(
    const bf16* __restrict__ Z, const bf16* __restrict__ Wf,
    const float* __restrict__ bias, bf16* __restrict__ T) {
  GEMM_PROLOG(Z, WIDTH, Wf, WIDTH >> 5)
  GEMM_KLOOP()
  const int colL = lane & 15;
  const int mB   = (lane < 16) ? 0 : 8;
  #pragma unroll
  for (int ms = 0; ms < 2; ++ms)
    #pragma unroll
    for (int ns = 0; ns < 4; ++ns) {
      int col = col0 + ns * 16 + colL;
      float bv = bias[col];
      #pragma unroll
      for (int r = 0; r < 8; ++r) {
        int row = row0 + ms * 16 + mB + r;
        float v = acc[ms][ns][r] + bv;
        v = v / (1.0f + __expf(-v));
        T[(size_t)row * HID + col] = f2bf(v);
      }
    }
}

// --- t @ w2 + b2, residual h += s*z: K=2048, N=1024 --------------------------
__global__ __launch_bounds__(256) void gemm_res_kernel(
    const bf16* __restrict__ T, const bf16* __restrict__ Wf,
    const float* __restrict__ bias, const float* __restrict__ res_scale,
    int blk, float* __restrict__ H) {
  GEMM_PROLOG(T, HID, Wf, HID >> 5)
  GEMM_KLOOP()
  const float s = res_scale[blk];
  const int colL = lane & 15;
  const int mB   = (lane < 16) ? 0 : 8;
  #pragma unroll
  for (int ms = 0; ms < 2; ++ms)
    #pragma unroll
    for (int ns = 0; ns < 4; ++ns) {
      int col = col0 + ns * 16 + colL;
      float bv = bias[col];
      #pragma unroll
      for (int r = 0; r < 8; ++r) {
        int row = row0 + ms * 16 + mB + r;
        size_t o = (size_t)row * WIDTH + col;
        H[o] = H[o] + s * (acc[ms][ns][r] + bv);
      }
    }
}

// ---------------------------------------------------------------------------
// Head: per-class gathered-row GEMM  h[rows] @ Wh[cls] + bh[cls]
// K=1024, N=256; rows gathered via LDS index tile, async-staged weights.
// ---------------------------------------------------------------------------
__global__ __launch_bounds__(256) void head_kernel(
    const float* __restrict__ H, const bf16* __restrict__ WhF,
    const float* __restrict__ bh, const int* __restrict__ cnt,
    const int* __restrict__ rows, float* __restrict__ out) {
  const int cls = blockIdx.z;
  const int nrows = cnt[cls];
  const int tileM = blockIdx.y * 256;
  if (tileM >= nrows) return;

  __shared__ __align__(32) bf16 Bs[2][2048];
  __shared__ int rlds[256];
  const int tid  = threadIdx.x;
  const int lane = tid & 31;
  const int wave = tid >> 5;
  const int nk   = WIDTH >> 5;  // 32

  {
    int idx = tileM + tid;
    rlds[tid] = (idx < nrows) ? rows[cls * B_TOT + idx] : -1;
  }

  const int col0 = blockIdx.x * 64;
  const int tn0  = blockIdx.x * 4;
  const int sidx = (tid >> 6) * 512 + (tid & 63) * 8;
  const bf16* Wf = WhF + (size_t)cls * (OUTD >> 4) * nk * 512;
  const bf16* gsrc = Wf + ((size_t)(tn0 + (tid >> 6)) * nk) * 512 +
                     (size_t)(tid & 63) * 8;
  async_stage_b128(gsrc, &Bs[0][sidx]);
  async_wait_barrier();   // also publishes rlds

  const int rloc0 = wave * 32;
  const int m  = lane & 15;
  const int kb = (lane >> 4) << 3;
  int rg0 = rlds[rloc0 + m];      if (rg0 < 0) rg0 = 0;
  int rg1 = rlds[rloc0 + 16 + m]; if (rg1 < 0) rg1 = 0;
  const float* pa0 = H + (size_t)rg0 * WIDTH + kb;
  const float* pa1 = H + (size_t)rg1 * WIDTH + kb;

  v8f acc[2][4] = {};
  for (int t = 0; t < nk; ++t) {
    const int buf = t & 1;
    if (t + 1 < nk)
      async_stage_b128(gsrc + (size_t)(t + 1) * 512, &Bs[buf ^ 1][sidx]);
    int k0 = t << 5;
    v16bf a0, a1;
    #pragma unroll
    for (int e = 0; e < 8; ++e) {
      a0[e]     = f2bf(pa0[k0 + e]);
      a0[e + 8] = f2bf(pa0[k0 + 16 + e]);
      a1[e]     = f2bf(pa1[k0 + e]);
      a1[e + 8] = f2bf(pa1[k0 + 16 + e]);
    }
    v16bf bfr[4];
    #pragma unroll
    for (int ns = 0; ns < 4; ++ns) bfr[ns] = load_frag_b(&Bs[buf][ns * 512], lane);
    #pragma unroll
    for (int ns = 0; ns < 4; ++ns) {
      acc[0][ns] = wmma_bf16(a0, bfr[ns], acc[0][ns]);
      acc[1][ns] = wmma_bf16(a1, bfr[ns], acc[1][ns]);
    }
    async_wait_barrier();
  }
  const int colL = lane & 15;
  const int mB   = (lane < 16) ? 0 : 8;
  #pragma unroll
  for (int ms = 0; ms < 2; ++ms)
    #pragma unroll
    for (int ns = 0; ns < 4; ++ns) {
      int col = col0 + ns * 16 + colL;
      float bv = bh[cls * OUTD + col];
      #pragma unroll
      for (int r = 0; r < 8; ++r) {
        int rl = rloc0 + ms * 16 + mB + r;
        int gr = rlds[rl];
        if (gr >= 0) out[(size_t)gr * OUTD + col] = acc[ms][ns][r] + bv;
      }
    }
}

// ---------------------------------------------------------------------------
// LayerNorm: one wave per row (1024 features)
// ---------------------------------------------------------------------------
__global__ __launch_bounds__(256) void ln_kernel(
    const float* __restrict__ H, const float* __restrict__ g,
    const float* __restrict__ b, bf16* __restrict__ Z) {
  const int lane = threadIdx.x & 31;
  const int wave = threadIdx.x >> 5;
  const int row  = blockIdx.x * 8 + wave;
  const float* h = H + (size_t)row * WIDTH;
  float vals[32];
  float s = 0.0f;
  #pragma unroll
  for (int j = 0; j < 32; ++j) { vals[j] = h[j * 32 + lane]; s += vals[j]; }
  #pragma unroll
  for (int off = 16; off > 0; off >>= 1) s += __shfl_xor(s, off, 32);
  float mu = s * (1.0f / WIDTH);
  float q = 0.0f;
  #pragma unroll
  for (int j = 0; j < 32; ++j) { float d = vals[j] - mu; q += d * d; }
  #pragma unroll
  for (int off = 16; off > 0; off >>= 1) q += __shfl_xor(q, off, 32);
  float rs = rsqrtf(q * (1.0f / WIDTH) + 1e-5f);
  bf16* z = Z + (size_t)row * WIDTH;
  #pragma unroll
  for (int j = 0; j < 32; ++j) {
    int c = j * 32 + lane;
    z[c] = f2bf((vals[j] - mu) * rs * g[c] + b[c]);
  }
}

// ---------------------------------------------------------------------------
// launch
// ---------------------------------------------------------------------------
extern "C" void kernel_launch(void* const* d_in, const int* in_sizes, int n_in,
                              void* d_out, int out_size, void* d_ws, size_t ws_size,
                              hipStream_t stream) {
  const float* theta = (const float*)d_in[0];
  const float* lam   = (const float*)d_in[1];
  const float* yoh   = (const float*)d_in[2];
  const float* W_in  = (const float*)d_in[3];
  const float* b_in  = (const float*)d_in[4];
  const float* ln_g  = (const float*)d_in[5];
  const float* ln_b  = (const float*)d_in[6];
  const float* w1    = (const float*)d_in[7];
  const float* b1    = (const float*)d_in[8];
  const float* w2    = (const float*)d_in[9];
  const float* b2    = (const float*)d_in[10];
  const float* rsc   = (const float*)d_in[11];
  const float* Wh    = (const float*)d_in[12];
  const float* bh    = (const float*)d_in[13];
  float* out = (float*)d_out;

  size_t off = 0;
  auto carve = [&](size_t bytes) -> char* {
    char* p = (char*)d_ws + off;
    off += (bytes + 255) & ~(size_t)255;
    return p;
  };
  const size_t F_IN = (size_t)(WIDTH / 16) * (KIN_PAD / 32) * 512;
  const size_t F_W1 = (size_t)(HID / 16)   * (WIDTH / 32)  * 512;
  const size_t F_W2 = (size_t)(WIDTH / 16) * (HID / 32)    * 512;
  const size_t F_WH = (size_t)(OUTD / 16)  * (WIDTH / 32)  * 512;

  bf16* x_bf  = (bf16*)carve((size_t)B_TOT * KIN_PAD * 2);
  bf16* winF  = (bf16*)carve(F_IN * 2);
  bf16* w1F   = (bf16*)carve(NBLK * F_W1 * 2);
  bf16* w2F   = (bf16*)carve(NBLK * F_W2 * 2);
  bf16* whF   = (bf16*)carve(YDIM * F_WH * 2);
  float* Hbuf = (float*)carve((size_t)B_TOT * WIDTH * 4);
  bf16* Zbuf  = (bf16*)carve((size_t)B_TOT * WIDTH * 2);
  bf16* Tbuf  = (bf16*)carve((size_t)B_TOT * HID * 2);
  int* cnt    = (int*)carve(YDIM * 4);
  int* rowsb  = (int*)carve((size_t)YDIM * B_TOT * 4);
  (void)ws_size; (void)n_in; (void)in_sizes; (void)out_size;

  build_x_kernel<<<(B_TOT * KIN_PAD + 255) / 256, 256, 0, stream>>>(theta, lam, yoh, x_bf);
  zero16_kernel<<<1, 64, 0, stream>>>(cnt);
  argmax_bin_kernel<<<(B_TOT + 255) / 256, 256, 0, stream>>>(yoh, cnt, rowsb);

  shuffle_w_kernel<<<(unsigned)((F_IN + 255) / 256), 256, 0, stream>>>(
      W_in, winF, KIN_REAL, KIN_PAD, WIDTH);
  for (int i = 0; i < NBLK; ++i) {
    shuffle_w_kernel<<<(unsigned)((F_W1 + 255) / 256), 256, 0, stream>>>(
        w1 + (size_t)i * WIDTH * HID, w1F + i * F_W1, WIDTH, WIDTH, HID);
    shuffle_w_kernel<<<(unsigned)((F_W2 + 255) / 256), 256, 0, stream>>>(
        w2 + (size_t)i * HID * WIDTH, w2F + i * F_W2, HID, HID, WIDTH);
  }
  for (int k = 0; k < YDIM; ++k) {
    shuffle_w_kernel<<<(unsigned)((F_WH + 255) / 256), 256, 0, stream>>>(
        Wh + (size_t)k * WIDTH * OUTD, whF + k * F_WH, WIDTH, WIDTH, OUTD);
  }

  dim3 blk(256);
  gemm_in_gelu_kernel<<<dim3(WIDTH / 64, B_TOT / 256), blk, 0, stream>>>(
      x_bf, winF, b_in, Hbuf);

  for (int i = 0; i < NBLK; ++i) {
    ln_kernel<<<B_TOT / 8, blk, 0, stream>>>(
        Hbuf, ln_g + (size_t)i * WIDTH, ln_b + (size_t)i * WIDTH, Zbuf);
    gemm_silu_kernel<<<dim3(HID / 64, B_TOT / 256), blk, 0, stream>>>(
        Zbuf, w1F + (size_t)i * F_W1, b1 + (size_t)i * HID, Tbuf);
    gemm_res_kernel<<<dim3(WIDTH / 64, B_TOT / 256), blk, 0, stream>>>(
        Tbuf, w2F + (size_t)i * F_W2, b2 + (size_t)i * WIDTH, rsc, i, Hbuf);
  }

  head_kernel<<<dim3(OUTD / 64, B_TOT / 256, YDIM), blk, 0, stream>>>(
      Hbuf, whF, bh, cnt, rowsb, out);
}
